// Classifier_24532853195314
// MI455X (gfx1250) — matrix operand order repaired and checked by reading, not compiled, not run on hardware
//
#include <hip/hip_runtime.h>

#define NROWS 65536
#define AROWS 1024
#define DDIM  768
#define KT    32
#define NT    (DDIM / KT)   // 24 k-steps
#define LDSW  40            // halves per LDS row: 32 data + 8 pad -> 80B stride, conflict-free

typedef __attribute__((ext_vector_type(16))) __bf16 v16bf;
typedef __attribute__((ext_vector_type(8)))  float  v8f;

__device__ __forceinline__ unsigned short f2bf(float f) {
    // round-to-nearest-even float -> bf16
    unsigned int u = __float_as_uint(f);
    u += 0x7FFFu + ((u >> 16) & 1u);
    return (unsigned short)(u >> 16);
}

// CDNA5 async copy: global -> LDS, 16B per lane, tracked by ASYNCcnt.
__device__ __forceinline__ void async_b128(unsigned lds_addr, const void* gaddr) {
    asm volatile("global_load_async_to_lds_b128 %0, %1, off"
                 :
                 : "v"(lds_addr), "v"(gaddr)
                 : "memory");
}

__device__ __forceinline__ void wait_async0() {
    asm volatile("s_wait_asynccnt 0x0" ::: "memory");
}

// One wave32 per row: load 768 f32, L2-normalize (norm clamped at 1e-8), store bf16.
__global__ __launch_bounds__(256) void rownorm_bf16(const float* __restrict__ src,
                                                    unsigned short* __restrict__ dst,
                                                    int rows) {
    const int lane = threadIdx.x & 31;
    const int wid  = threadIdx.x >> 5;
    const int row  = blockIdx.x * 8 + wid;
    if (row >= rows) return;
    const float* s = src + (size_t)row * DDIM;
    float v[DDIM / 32];
    float ss = 0.0f;
#pragma unroll
    for (int i = 0; i < DDIM / 32; ++i) { v[i] = s[lane + i * 32]; ss += v[i] * v[i]; }
#pragma unroll
    for (int m = 16; m >= 1; m >>= 1) ss += __shfl_xor(ss, m, 32);
    const float inv = 1.0f / fmaxf(sqrtf(ss), 1e-8f);
    unsigned short* d = dst + (size_t)row * DDIM;
#pragma unroll
    for (int i = 0; i < DDIM / 32; ++i) d[lane + i * 32] = f2bf(v[i] * inv);
}

// C[n,a] = dot(Ehat[n,:], What[a,:]) ; out = 10*C + 10.0000002
// Block tile 128(M) x 256(N), K-step 32, double-buffered async LDS staging.
// 8 waves as 2(M) x 4(N); each wave owns 64x64 = 4x4 WMMA tiles (16 wmma / k-step).
__global__ __launch_bounds__(256) void cosgemm(const unsigned short* __restrict__ E,
                                               const unsigned short* __restrict__ W,
                                               float* __restrict__ out) {
    __shared__ __align__(16) unsigned short sA[2][128 * LDSW];  // 2 x 10240 B
    __shared__ __align__(16) unsigned short sB[2][256 * LDSW];  // 2 x 20480 B

    const int t     = threadIdx.x;
    const int lane  = t & 31;
    const int w     = t >> 5;
    const int wm    = w & 1;         // 0..1 -> 64-row span in M
    const int wn    = w >> 1;        // 0..3 -> 64-col span in N
    const int laneM = lane & 15;     // row/col within 16x16 tile
    const int laneH = lane >> 4;     // half-wave selector

    const size_t mBase = (size_t)blockIdx.y * 128;
    const size_t aBase = (size_t)blockIdx.x * 256;

    const unsigned short* Eb = E + mBase * DDIM;
    const unsigned short* Wb = W + aBase * DDIM;

    // cooperative async tile load: 16B chunk per lane; 64B per k-row, row stride 1536B in global
    const int seg   = t & 3;   // which 16B chunk of the 64B k-row
    const int rbase = t >> 2;  // 0..63

    const unsigned sAbase = (unsigned)(size_t)&sA[0][0];
    const unsigned sBbase = (unsigned)(size_t)&sB[0][0];

    auto issue_tile = [&](int kt, int buf) {
        const int k0 = kt * KT;
        const unsigned aB = sAbase + (unsigned)buf * (128 * LDSW * 2);
        const unsigned bB = sBbase + (unsigned)buf * (256 * LDSW * 2);
#pragma unroll
        for (int i = 0; i < 2; ++i) {       // A: 128 rows
            const int row = rbase + i * 64;
            async_b128(aB + (unsigned)(row * LDSW + seg * 8) * 2,
                       Eb + (size_t)row * DDIM + k0 + seg * 8);
        }
#pragma unroll
        for (int i = 0; i < 4; ++i) {       // B: 256 rows
            const int row = rbase + i * 64;
            async_b128(bB + (unsigned)(row * LDSW + seg * 8) * 2,
                       Wb + (size_t)row * DDIM + k0 + seg * 8);
        }
    };

    const v8f vzero = {0.f, 0.f, 0.f, 0.f, 0.f, 0.f, 0.f, 0.f};
    v8f acc[4][4];
#pragma unroll
    for (int mi = 0; mi < 4; ++mi)
#pragma unroll
        for (int nj = 0; nj < 4; ++nj) acc[mi][nj] = vzero;

    issue_tile(0, 0);

#pragma unroll 1
    for (int kt = 0; kt < NT; ++kt) {
        const int cur = kt & 1;
        wait_async0();        // my async writes into buffer `cur` have landed
        __syncthreads();      // everyone's writes landed; everyone done reading buffer cur^1
        if (kt + 1 < NT) issue_tile(kt + 1, cur ^ 1);  // overlaps the WMMA phase below

        const unsigned short* cA = &sA[cur][0];
        const unsigned short* cB = &sB[cur][0];

        // A fragments: lane L -> row M=L%16; K {0..7,16..23} (lanes 0-15) / {8..15,24..31} (16-31)
        // B fragments: lane L -> col N=L%16; K {0..7,16..23} (lanes 0-15) / {8..15,24..31} (16-31)
        union Frag { uint4 u[2]; v16bf v; } a[4], b[4];
#pragma unroll
        for (int mi = 0; mi < 4; ++mi) {
            const int rr = wm * 64 + mi * 16 + laneM;
            a[mi].u[0] = *(const uint4*)&cA[rr * LDSW + laneH * 8];
            a[mi].u[1] = *(const uint4*)&cA[rr * LDSW + 16 + laneH * 8];
        }
#pragma unroll
        for (int nj = 0; nj < 4; ++nj) {
            const int rr = wn * 64 + nj * 16 + laneM;
            b[nj].u[0] = *(const uint4*)&cB[rr * LDSW + laneH * 8];
            b[nj].u[1] = *(const uint4*)&cB[rr * LDSW + 16 + laneH * 8];
        }
#pragma unroll
        for (int mi = 0; mi < 4; ++mi)
#pragma unroll
            for (int nj = 0; nj < 4; ++nj)
                acc[mi][nj] = __builtin_amdgcn_wmma_f32_16x16x32_bf16(
                    false, a[mi].v, false, b[nj].v, (short)0, acc[mi][nj], false, false);
    }

    // epilogue: D layout -> VGPR v: M = v + 8*laneH, N = laneM
#pragma unroll
    for (int mi = 0; mi < 4; ++mi) {
#pragma unroll
        for (int nj = 0; nj < 4; ++nj) {
            const size_t gr = mBase + (size_t)(wm * 64 + mi * 16 + laneH * 8);
            const size_t gc = aBase + (size_t)(wn * 64 + nj * 16 + laneM);
            float* o = out + gr * AROWS + gc;
#pragma unroll
            for (int v = 0; v < 8; ++v)
                o[(size_t)v * AROWS] = fmaf(acc[mi][nj][v], 10.0f, 10.0000002f);
        }
    }
}

extern "C" void kernel_launch(void* const* d_in, const int* in_sizes, int n_in,
                              void* d_out, int out_size, void* d_ws, size_t ws_size,
                              hipStream_t stream) {
    (void)in_sizes; (void)n_in; (void)out_size; (void)ws_size;
    const float* emb    = (const float*)d_in[0];
    const float* weight = (const float*)d_in[1];
    float* out          = (float*)d_out;

    unsigned short* embN = (unsigned short*)d_ws;            // 65536*768 bf16
    unsigned short* wN   = embN + (size_t)NROWS * DDIM;      // 1024*768 bf16

    rownorm_bf16<<<NROWS / 8, 256, 0, stream>>>(emb, embN, NROWS);
    rownorm_bf16<<<AROWS / 8, 256, 0, stream>>>(weight, wN, AROWS);

    dim3 grid(AROWS / 256, NROWS / 128);  // (4, 512)
    cosgemm<<<grid, 256, 0, stream>>>(embN, wN, out);
}
